// HeteroClassifier_16810501997194
// MI455X (gfx1250) — compile-verified, matrix operand order checked by reading, not run on hardware
//
#include <hip/hip_runtime.h>
#include <hip/hip_bf16.h>
#include <math.h>

// ---------------- problem constants (match reference) ----------------
#define NB   128        // news nodes / graphs
#define NUU  128000     // user nodes
#define NE   256000     // edges per relation
#define DIN  300
#define HH   256
#define UPG  1000

// ---------------- WMMA types ----------------
typedef __attribute__((ext_vector_type(16))) __bf16         v16bf;
typedef __attribute__((ext_vector_type(8)))  float          v8f;
typedef __attribute__((ext_vector_type(8)))  unsigned short ushort8;
typedef __attribute__((ext_vector_type(16))) unsigned short ushort16;

__device__ __forceinline__ unsigned short f2bf(float f) {
  unsigned u = __builtin_bit_cast(unsigned, f);
  unsigned r = u + 0x7FFFu + ((u >> 16) & 1u);   // round-to-nearest-even
  return (unsigned short)(r >> 16);
}
__device__ __forceinline__ float bf2f(unsigned short h) {
  return __builtin_bit_cast(float, ((unsigned)h) << 16);
}

// ---- gfx1250 async copy: global -> LDS, 16 bytes, tracked by ASYNCcnt ----
__device__ __forceinline__ void async_copy_b128(unsigned lds_off, const void* g) {
  asm volatile("global_load_async_to_lds_b128 %0, %1, off"
               :: "v"(lds_off), "v"((unsigned long long)(size_t)g)
               : "memory");
}
__device__ __forceinline__ void wait_async0() {   // all transfers done
  asm volatile("s_wait_asynccnt 0x0" ::: "memory");
}
__device__ __forceinline__ void wait_async3() {   // previous tile (3 xfers) done,
  asm volatile("s_wait_asynccnt 0x3" ::: "memory"); // next tile's 3 still in flight
}

// ---------------- generic WMMA GEMM ----------------
// C[M,N] = op(Ab[M,Kp](bf16) @ WT[Np,Kp](bf16, pre-transposed)) ; Kp%32==0, M%128==0
// FLAGS: bit0 relu, bit1 accumulate into C, bit2 output bf16 (ld = N)
#define TM 128
#define TN 64
#define TK 32
#define KS 48    // LDS row stride (bf16 elems): 96B -> every 8-elem chunk 16B aligned
#define F_RELU  1
#define F_ACC   2
#define F_OBF16 4

template<int FLAGS>
__global__ __launch_bounds__(256)
void gemm_wmma_bf16(const unsigned short* __restrict__ Ab,
                    const unsigned short* __restrict__ WT,
                    const float* __restrict__ bias,
                    void* __restrict__ Cout,
                    int N, int Kp)
{
  __shared__ __align__(16) unsigned short lA[2][TM * KS]; // ping-pong [m][k]
  __shared__ __align__(16) unsigned short lB[2][TN * KS]; // ping-pong [n][k]

  const int tx   = threadIdx.x;
  const int wave = tx >> 5;
  const int lane = tx & 31;
  const int mBlk = blockIdx.y * TM;
  const int nBlk = blockIdx.x * TN;

  v8f acc[4];
  acc[0] = v8f{}; acc[1] = v8f{}; acc[2] = v8f{}; acc[3] = v8f{};

  const int rsel  = lane & 15;          // row (A) / col (B,C) within 16-tile
  const int khalf = (lane >> 4) << 3;   // 0 or 8 per ISA 16-bit operand layout

  // per-thread async chunk coordinates (8 bf16 = 16B per transfer)
  const int rc = tx >> 2;               // 0..63
  const int cc = (tx & 3) * 8;          // 0,8,16,24

  // stage tile at k-offset k0 into ping-pong slot p (3 b128 asyncs per thread)
  auto stage = [&](int k0, int p) {
    unsigned a0 = (unsigned)(size_t)&lA[p][0];
    unsigned b0 = (unsigned)(size_t)&lB[p][0];
    async_copy_b128(a0 + (unsigned)(rc * KS + cc) * 2,
                    Ab + (size_t)(mBlk + rc) * Kp + k0 + cc);
    async_copy_b128(a0 + (unsigned)((rc + 64) * KS + cc) * 2,
                    Ab + (size_t)(mBlk + rc + 64) * Kp + k0 + cc);
    async_copy_b128(b0 + (unsigned)(rc * KS + cc) * 2,
                    WT + (size_t)(nBlk + rc) * Kp + k0 + cc);
  };

  stage(0, 0);                                    // prologue: prefetch tile 0
  int p = 0;
  for (int k0 = 0; k0 < Kp; k0 += TK, p ^= 1) {
    // software pipeline: issue tile t+1 into other slot, then retire tile t only
    if (k0 + TK < Kp) { stage(k0 + TK, p ^ 1); wait_async3(); }
    else              { wait_async0(); }
    __syncthreads();                              // tile t visible to all waves

    // A fragment: lane holds row (lane&15); K = {khalf..+7, 16+khalf..+7}
    const unsigned short* ap = &lA[p][(wave * 16 + rsel) * KS];
    ushort8 alo = *(const ushort8*)(ap + khalf);
    ushort8 ahi = *(const ushort8*)(ap + 16 + khalf);
    ushort16 au = __builtin_shufflevector(alo, ahi, 0,1,2,3,4,5,6,7,8,9,10,11,12,13,14,15);
    v16bf afrag = __builtin_bit_cast(v16bf, au);

    #pragma unroll
    for (int j = 0; j < 4; ++j) {
      const unsigned short* bp = &lB[p][(j * 16 + rsel) * KS];
      ushort8 blo = *(const ushort8*)(bp + khalf);
      ushort8 bhi = *(const ushort8*)(bp + 16 + khalf);
      ushort16 bu = __builtin_shufflevector(blo, bhi, 0,1,2,3,4,5,6,7,8,9,10,11,12,13,14,15);
      v16bf bfrag = __builtin_bit_cast(v16bf, bu);
      acc[j] = __builtin_amdgcn_wmma_f32_16x16x32_bf16(
          false, afrag, false, bfrag, (short)0, acc[j], false, false);
    }
    __syncthreads();                              // all waves done reading slot p
  }

  // epilogue: C layout per ISA (VGPR r -> row r + khalf, col = lane&15)
  const int rowb = mBlk + wave * 16 + khalf;
  float* Cf = (float*)Cout;
  unsigned short* Cb = (unsigned short*)Cout;
  #pragma unroll
  for (int j = 0; j < 4; ++j) {
    int gn = nBlk + j * 16 + rsel;
    if (gn < N) {
      #pragma unroll
      for (int r = 0; r < 8; ++r) {
        size_t idx = (size_t)(rowb + r) * N + gn;  // M % 128 == 0 guaranteed
        float v = acc[j][r];
        if (bias) v += bias[gn];
        if constexpr (FLAGS & F_ACC)
          v += (FLAGS & F_OBF16) ? bf2f(Cb[idx]) : Cf[idx];
        if constexpr (FLAGS & F_RELU) v = fmaxf(v, 0.0f);
        if constexpr (FLAGS & F_OBF16) Cb[idx] = f2bf(v); else Cf[idx] = v;
      }
    }
  }
}

// ---------------- f32 -> padded bf16 rows: dst[r, 0:Kp] (zero pad k>=K) ----------------
__global__ void cvt_pad_bf16(const float* __restrict__ src, unsigned short* __restrict__ dst,
                             int K, int Kp)
{
  int r = blockIdx.x;
  int k = blockIdx.y * blockDim.x + threadIdx.x;
  if (k < Kp)
    dst[(size_t)r * Kp + k] = (k < K) ? f2bf(src[(size_t)r * K + k]) : (unsigned short)0;
}

// ---------------- f32 W[K,N] -> bf16 WT[Np,Kp] transposed + zero-padded ----------------
__global__ void cvt_wt_bf16(const float* __restrict__ W, unsigned short* __restrict__ WT,
                            int K, int N, int Kp)
{
  int n = blockIdx.x;                  // < Np
  int k = blockIdx.y * blockDim.x + threadIdx.x;
  if (k < Kp)
    WT[(size_t)n * Kp + k] = (n < N && k < K) ? f2bf(W[(size_t)k * N + n]) : (unsigned short)0;
}

// ---------------- zero fill ----------------
__global__ void fill_zero(float* __restrict__ p, size_t n) {
  size_t i = (size_t)blockIdx.x * blockDim.x + threadIdx.x;
  size_t stride = (size_t)gridDim.x * blockDim.x;
  for (; i < n; i += stride) p[i] = 0.0f;
}

// ---------------- edge scatter segment-max (relu inputs >= 0) ----------------
__global__ void edge_segmax(const float* __restrict__ m, const int* __restrict__ src,
                            const int* __restrict__ dst, unsigned* __restrict__ agg,
                            int F)
{
  int e = blockIdx.x;
  int s = src[e], d = dst[e];
  const float* mr = m + (size_t)s * F;
  unsigned*    ar = agg + (size_t)d * F;
  for (int f = threadIdx.x * 4; f < F; f += blockDim.x * 4) {
    float4 v = *(const float4*)(mr + f);
    atomicMax(&ar[f + 0], __float_as_uint(v.x));
    atomicMax(&ar[f + 1], __float_as_uint(v.y));
    atomicMax(&ar[f + 2], __float_as_uint(v.z));
    atomicMax(&ar[f + 3], __float_as_uint(v.w));
  }
}

// ---------------- per-graph mean readout -> bf16 hg0 ----------------
__global__ __launch_bounds__(HH)
void readout_mean(const float* __restrict__ h_user2, const float* __restrict__ h_news2,
                  unsigned short* __restrict__ hg0b)
{
  int g = blockIdx.x;
  int f = threadIdx.x;                 // blockDim == HH
  const float* base = h_user2 + (size_t)g * UPG * HH + f;
  float s = 0.0f;
  for (int u = 0; u < UPG; ++u) s += base[(size_t)u * HH];
  hg0b[(size_t)g * HH + f] = f2bf(h_news2[(size_t)g * HH + f] + s * (1.0f / UPG));
}

// ---------------- classifier + log_softmax ----------------
__global__ void classifier(const float* __restrict__ hg, const float* __restrict__ nh,
                           const float* __restrict__ clsW, const float* __restrict__ clsb,
                           float* __restrict__ out)
{
  int b = blockIdx.x * blockDim.x + threadIdx.x;
  if (b >= NB) return;
  float a0 = clsb[0], a1 = clsb[1];
  const float* hr = hg + (size_t)b * HH;
  const float* nr = nh + (size_t)b * HH;
  for (int j = 0; j < HH; ++j) { float x = hr[j]; a0 += x * clsW[j*2]; a1 += x * clsW[j*2+1]; }
  for (int j = 0; j < HH; ++j) { float x = nr[j]; a0 += x * clsW[(HH+j)*2]; a1 += x * clsW[(HH+j)*2+1]; }
  float mx  = fmaxf(a0, a1);
  float lse = mx + logf(expf(a0 - mx) + expf(a1 - mx));
  out[b*2+0] = a0 - lse;
  out[b*2+1] = a1 - lse;
}

// ---------------- host helpers ----------------
static inline int rup(int x, int m) { return (x + m - 1) / m * m; }

static inline void gemm(const unsigned short* A, const unsigned short* WT, const float* bias,
                        void* C, int M, int N, int Kp, int flags, hipStream_t s)
{
  dim3 g((N + TN - 1) / TN, (M + TM - 1) / TM);
  dim3 b(256);
  switch (flags) {
    case 0:                          gemm_wmma_bf16<0><<<g, b, 0, s>>>(A, WT, bias, C, N, Kp); break;
    case F_RELU:                     gemm_wmma_bf16<F_RELU><<<g, b, 0, s>>>(A, WT, bias, C, N, Kp); break;
    case F_ACC:                      gemm_wmma_bf16<F_ACC><<<g, b, 0, s>>>(A, WT, bias, C, N, Kp); break;
    case F_ACC | F_RELU:             gemm_wmma_bf16<F_ACC | F_RELU><<<g, b, 0, s>>>(A, WT, bias, C, N, Kp); break;
    case F_OBF16:                    gemm_wmma_bf16<F_OBF16><<<g, b, 0, s>>>(A, WT, bias, C, N, Kp); break;
    default:                         gemm_wmma_bf16<F_OBF16 | F_ACC | F_RELU><<<g, b, 0, s>>>(A, WT, bias, C, N, Kp); break;
  }
}
static inline void cvt(const float* src, unsigned short* dst, int rows, int K, int Kp,
                       hipStream_t s)
{
  cvt_pad_bf16<<<dim3(rows, (Kp + 255) / 256), 256, 0, s>>>(src, dst, K, Kp);
}
static inline void wtc(const float* W, unsigned short* WT, int K, int N, hipStream_t s)
{
  cvt_wt_bf16<<<dim3(rup(N, TN), (rup(K, TK) + 255) / 256), 256, 0, s>>>(W, WT, K, N, rup(K, TK));
}
static inline void zfill(float* p, size_t n, hipStream_t s) {
  fill_zero<<<2048, 256, 0, s>>>(p, n);
}

extern "C" void kernel_launch(void* const* d_in, const int* in_sizes, int n_in,
                              void* d_out, int out_size, void* d_ws, size_t ws_size,
                              hipStream_t stream)
{
  (void)in_sizes; (void)n_in; (void)out_size; (void)ws_size;
  const float* news_feat = (const float*)d_in[0];
  const float* user_feat = (const float*)d_in[1];
  const int*   e_pn_src  = (const int*)d_in[2];
  const int*   e_pn_dst  = (const int*)d_in[3];
  const int*   e_np_src  = (const int*)d_in[4];
  const int*   e_np_dst  = (const int*)d_in[5];
  const float* Wp1pn=(const float*)d_in[6],  *bp1pn=(const float*)d_in[7];
  const float* Ws1pn=(const float*)d_in[8],  *Wn1pn=(const float*)d_in[9],  *b1pn=(const float*)d_in[10];
  const float* Wp1np=(const float*)d_in[11], *bp1np=(const float*)d_in[12];
  const float* Ws1np=(const float*)d_in[13], *Wn1np=(const float*)d_in[14], *b1np=(const float*)d_in[15];
  const float* Wp2pn=(const float*)d_in[16], *bp2pn=(const float*)d_in[17];
  const float* Ws2pn=(const float*)d_in[18], *Wn2pn=(const float*)d_in[19], *b2pn=(const float*)d_in[20];
  const float* Wp2np=(const float*)d_in[21], *bp2np=(const float*)d_in[22];
  const float* Ws2np=(const float*)d_in[23], *Wn2np=(const float*)d_in[24], *b2np=(const float*)d_in[25];
  const float* lin1W=(const float*)d_in[26], *lin1b=(const float*)d_in[27];
  const float* lin2W=(const float*)d_in[28], *lin2b=(const float*)d_in[29];
  const float* clsW =(const float*)d_in[30], *clsb =(const float*)d_in[31];

  const int KpD = rup(DIN, TK);   // 320
  const int KpH = HH;             // 256

  // ---- bump allocator on workspace (256B aligned) ----
  char* base = (char*)d_ws;
  size_t off = 0;
  auto alloc = [&](size_t bytes) -> void* {
    void* p = base + off;
    off = (off + bytes + 255) & ~(size_t)255;
    return p;
  };
  // f32 buffers
  float* bufA = (float*)alloc((size_t)NUU * DIN * 4);  // m1_user / m2_user / h_user2
  float* bufB = (float*)alloc((size_t)NUU * DIN * 4);  // agg_user1 / agg_user2
  float* agg_news1 = (float*)alloc((size_t)NB * DIN * 4);
  float* m1_news   = (float*)alloc((size_t)NB * DIN * 4);
  float* agg_news2 = (float*)alloc((size_t)NB * HH * 4);
  float* m2_news   = (float*)alloc((size_t)NB * HH * 4);
  float* h_news2   = (float*)alloc((size_t)NB * HH * 4);
  float* hg        = (float*)alloc((size_t)NB * HH * 4);
  float* newsh     = (float*)alloc((size_t)NB * HH * 4);
  // bf16 buffers (padded to Kp)
  unsigned short* regA = (unsigned short*)alloc((size_t)NUU * KpD * 2); // xb_user
  unsigned short* regB = (unsigned short*)alloc((size_t)NUU * KpD * 2); // aggb_user1/2
  unsigned short* regC = (unsigned short*)alloc((size_t)NUU * KpH * 2); // hb_user1
  unsigned short* xb_news    = (unsigned short*)alloc((size_t)NB * KpD * 2);
  unsigned short* aggb_news1 = (unsigned short*)alloc((size_t)NB * KpD * 2);
  unsigned short* hb_news1   = (unsigned short*)alloc((size_t)NB * KpH * 2);
  unsigned short* aggb_news2 = (unsigned short*)alloc((size_t)NB * KpH * 2);
  unsigned short* hg0b       = (unsigned short*)alloc((size_t)NB * KpH * 2);
  // pre-transposed bf16 weights [Np, Kp]
  auto walloc = [&](int K, int N) { return (unsigned short*)alloc((size_t)rup(N, TN) * rup(K, TK) * 2); };
  unsigned short *WTp1pn = walloc(DIN, DIN), *WTs1pn = walloc(DIN, HH), *WTn1pn = walloc(DIN, HH);
  unsigned short *WTp1np = walloc(DIN, DIN), *WTs1np = walloc(DIN, HH), *WTn1np = walloc(DIN, HH);
  unsigned short *WTp2pn = walloc(HH, HH),   *WTs2pn = walloc(HH, HH),  *WTn2pn = walloc(HH, HH);
  unsigned short *WTp2np = walloc(HH, HH),   *WTs2np = walloc(HH, HH),  *WTn2np = walloc(HH, HH);
  unsigned short *WTlin1 = walloc(HH, HH),   *WTlin2 = walloc(DIN, HH);

  // ---- one-time-per-call conversions ----
  cvt(user_feat, regA,    NUU, DIN, KpD, stream);      // xb_user
  cvt(news_feat, xb_news, NB,  DIN, KpD, stream);
  wtc(Wp1pn, WTp1pn, DIN, DIN, stream);  wtc(Ws1pn, WTs1pn, DIN, HH, stream);
  wtc(Wn1pn, WTn1pn, DIN, HH, stream);   wtc(Wp1np, WTp1np, DIN, DIN, stream);
  wtc(Ws1np, WTs1np, DIN, HH, stream);   wtc(Wn1np, WTn1np, DIN, HH, stream);
  wtc(Wp2pn, WTp2pn, HH, HH, stream);    wtc(Ws2pn, WTs2pn, HH, HH, stream);
  wtc(Wn2pn, WTn2pn, HH, HH, stream);    wtc(Wp2np, WTp2np, HH, HH, stream);
  wtc(Ws2np, WTs2np, HH, HH, stream);    wtc(Wn2np, WTn2np, HH, HH, stream);
  wtc(lin1W, WTlin1, HH, HH, stream);    wtc(lin2W, WTlin2, DIN, HH, stream);

  // ---------- layer 1, User->News ----------
  gemm(regA, WTp1pn, bp1pn, bufA, NUU, DIN, KpD, F_RELU, stream);           // m1_user f32
  zfill(agg_news1, (size_t)NB * DIN, stream);
  edge_segmax<<<NE, 128, 0, stream>>>(bufA, e_pn_src, e_pn_dst, (unsigned*)agg_news1, DIN);
  // ---------- layer 1, News->User ----------
  gemm(xb_news, WTp1np, bp1np, m1_news, NB, DIN, KpD, F_RELU, stream);
  zfill(bufB, (size_t)NUU * DIN, stream);
  edge_segmax<<<NE, 128, 0, stream>>>(m1_news, e_np_src, e_np_dst, (unsigned*)bufB, DIN);
  cvt(agg_news1, aggb_news1, NB, DIN, KpD, stream);
  cvt(bufB, regB, NUU, DIN, KpD, stream);                                   // aggb_user1
  // h_news1 (bf16 out) = relu(news@Ws + agg@Wn + b)
  gemm(xb_news, WTs1pn, b1pn, hb_news1, NB, HH, KpD, F_OBF16, stream);
  gemm(aggb_news1, WTn1pn, nullptr, hb_news1, NB, HH, KpD, F_OBF16 | F_ACC | F_RELU, stream);
  // h_user1 (bf16 out)
  gemm(regA, WTs1np, b1np, regC, NUU, HH, KpD, F_OBF16, stream);
  gemm(regB, WTn1np, nullptr, regC, NUU, HH, KpD, F_OBF16 | F_ACC | F_RELU, stream);

  // ---------- layer 2, User->News ----------
  gemm(regC, WTp2pn, bp2pn, bufA, NUU, HH, KpH, F_RELU, stream);            // m2_user f32
  zfill(agg_news2, (size_t)NB * HH, stream);
  edge_segmax<<<NE, 128, 0, stream>>>(bufA, e_pn_src, e_pn_dst, (unsigned*)agg_news2, HH);
  // ---------- layer 2, News->User ----------
  gemm(hb_news1, WTp2np, bp2np, m2_news, NB, HH, KpH, F_RELU, stream);
  zfill(bufB, (size_t)NUU * HH, stream);
  edge_segmax<<<NE, 128, 0, stream>>>(m2_news, e_np_src, e_np_dst, (unsigned*)bufB, HH);
  cvt(agg_news2, aggb_news2, NB, HH, KpH, stream);
  cvt(bufB, regB, NUU, HH, KpH, stream);                                    // aggb_user2
  // h_news2 f32 (no relu)
  gemm(hb_news1, WTs2pn, b2pn, h_news2, NB, HH, KpH, 0, stream);
  gemm(aggb_news2, WTn2pn, nullptr, h_news2, NB, HH, KpH, F_ACC, stream);
  // h_user2 f32 -> bufA (m2_user dead)
  gemm(regC, WTs2np, b2np, bufA, NUU, HH, KpH, 0, stream);
  gemm(regB, WTn2np, nullptr, bufA, NUU, HH, KpH, F_ACC, stream);

  // ---------- readout + MLP + classifier ----------
  readout_mean<<<NB, HH, 0, stream>>>(bufA, h_news2, hg0b);
  gemm(hg0b, WTlin1, lin1b, hg, NB, HH, KpH, F_RELU, stream);
  gemm(xb_news, WTlin2, lin2b, newsh, NB, HH, KpD, F_RELU, stream);
  classifier<<<1, 128, 0, stream>>>(hg, newsh, clsW, clsb, (float*)d_out);
}
